// WindowAttention_21285857919703
// MI455X (gfx1250) — matrix operand order repaired and checked by reading, not compiled, not run on hardware
//
#include <hip/hip_runtime.h>
#include <hip/hip_bf16.h>

// ---------------- problem constants ----------------
#define DIMC   128
#define NHEADS 4
#define HD     32
#define NTOK   98
#define NPAD   112   // 7 * 16 (token dim padded for 16x16 tiles)
#define KPAD   128   // token dim padded for K=32 chunks in P*V
#define BWIN   2048
#define NWIN   64
#define NN     (NTOK * NTOK)   // 9604
// combined bias+mask table: [4][64][7][7][16][16] floats (12.8 MB)
#define COMB_ELEMS (NHEADS * NWIN * 7 * 7 * 256)

typedef __attribute__((ext_vector_type(16))) __bf16 v16bf;
typedef __attribute__((ext_vector_type(8)))  __bf16 v8bf;
typedef __attribute__((ext_vector_type(4)))  __bf16 v4bf;
typedef __attribute__((ext_vector_type(8)))  float  v8f;
typedef __attribute__((ext_vector_type(4)))  float  v4f;

union Frag16 { v16bf v; v8bf h[2]; uint4 u[2]; };
union BF4 { v4bf v; uint2 u; };

// ---------------------------------------------------------------
// Kernel 0: build padded combined (bias + mask) table in the WMMA
// C-fragment tile layout: comb[h][w][mi][t][col][row], row = e + 8*s.
// Padding (q>=98 or k>=98) = -1e30 so softmax masking needs no branches.
// ---------------------------------------------------------------
__global__ void __launch_bounds__(256) comb_kernel(
    const int* __restrict__ rpi, const float* __restrict__ table,
    const float* __restrict__ mask, float* __restrict__ comb) {
  int i = blockIdx.x * 256 + threadIdx.x;
  if (i >= COMB_ELEMS) return;
  int row = i & 15;
  int col = (i >> 4) & 15;
  int rest = i >> 8;
  int t = rest % 7;  rest /= 7;
  int mi = rest % 7; rest /= 7;
  int w = rest & 63;
  int h = rest >> 6;
  int q = 16 * mi + row;
  int k = 16 * t + col;
  float v = -1e30f;
  if (q < NTOK && k < NTOK) {
    int idx = rpi[q * NTOK + k];
    v = table[idx * NHEADS + h] + mask[(size_t)w * NN + q * NTOK + k];
  }
  comb[i] = v;
}

// ---------------------------------------------------------------
// Kernel 1: windowed attention, one block per (window, head)
// block = 256 threads = 8 wave32; waves 0..6 own 16-row tile slabs
// ---------------------------------------------------------------
__global__ void __launch_bounds__(256) attn_kernel(
    const float* __restrict__ x,      // (B_, 98, 384)
    const float* __restrict__ comb,   // (4, 64, 7, 7, 16, 16) padded bias+mask
    float* __restrict__ Obuf) {       // (B_, 98, 128)  head-concat layout
  __shared__ __align__(16) __bf16 Qs[NPAD * HD];     // 112x32
  __shared__ __align__(16) __bf16 Ks[NPAD * HD];     // 112x32
  __shared__ __align__(16) __bf16 Vts[HD * KPAD];    // 32x128 (V^T)
  __shared__ __align__(16) __bf16 Ps[NPAD * KPAD];   // 112x128

  const int b   = blockIdx.x;
  const int h   = blockIdx.y;
  const int tid = threadIdx.x;
  const float scale = 0.17677669529663687f;   // 32^-0.5

  const float* xw = x + (size_t)b * NTOK * (3 * DIMC);

  // ---- stage Q (pre-scaled) and K: coalesced v4f loads, packed bf16 cvt ----
  v4f zero4 = {0.f, 0.f, 0.f, 0.f};
  for (int i = tid; i < NPAD * HD / 4; i += 256) {   // 896 groups of 4
    int n = i >> 3, d4 = (i & 7) << 2;
    v4f qv = zero4, kv = zero4;
    if (n < NTOK) {
      const float* row = xw + (size_t)n * (3 * DIMC);
      qv = *(const v4f*)(row + h * HD + d4) * scale;
      kv = *(const v4f*)(row + DIMC + h * HD + d4);
    }
    BF4 qb, kb;
    qb.v = __builtin_convertvector(qv, v4bf);
    kb.v = __builtin_convertvector(kv, v4bf);
    *(uint2*)&Qs[i * 4] = qb.u;
    *(uint2*)&Ks[i * 4] = kb.u;
  }
  // ---- stage V^T: coalesced v4f loads, scatter bf16 into LDS columns ----
  for (int i = tid; i < KPAD * HD / 4; i += 256) {   // 1024 groups of 4
    int n = i >> 3, d4 = (i & 7) << 2;
    v4f vv = zero4;
    if (n < NTOK)
      vv = *(const v4f*)(xw + (size_t)n * (3 * DIMC) + 2 * DIMC + h * HD + d4);
    v4bf vb = __builtin_convertvector(vv, v4bf);
    Vts[(d4 + 0) * KPAD + n] = vb[0];
    Vts[(d4 + 1) * KPAD + n] = vb[1];
    Vts[(d4 + 2) * KPAD + n] = vb[2];
    Vts[(d4 + 3) * KPAD + n] = vb[3];
  }
  __syncthreads();

  const int wave = tid >> 5;
  if (wave >= 7) return;                 // 7 tile rows; wave 7 idles
  const int lane = tid & 31;
  const int n16  = lane & 15;
  const int s    = lane >> 4;            // half-wave select
  const int mi   = wave;                 // M-tile row index

  // per-lane base into the combined bias+mask table for this tile row
  const float* cb = comb +
      ((((size_t)h * NWIN + (b & (NWIN - 1))) * 7 + mi) * 7) * 256 +
      n16 * 16 + 8 * s;
  #pragma unroll
  for (int t = 0; t < 7; ++t)            // warm L2 while WMMAs run
    __builtin_prefetch((const void*)(cb + t * 256), 0, 1);

  // ---- S = (Q*scale) K^T : 7 WMMA ----
  Frag16 aQ;
  {
    int base = (16 * mi + n16) * HD;
    aQ.u[0] = *(const uint4*)&Qs[base + 8 * s];
    aQ.u[1] = *(const uint4*)&Qs[base + 16 + 8 * s];
  }
  v8f zero8 = {0.f, 0.f, 0.f, 0.f, 0.f, 0.f, 0.f, 0.f};
  v8f acc[7];
  #pragma unroll
  for (int t = 0; t < 7; ++t) acc[t] = zero8;
  #pragma unroll
  for (int t = 0; t < 7; ++t) {
    Frag16 bK;
    int base = (16 * t + n16) * HD + 8 * s;
    bK.u[0] = *(const uint4*)&Ks[base];
    bK.u[1] = *(const uint4*)&Ks[base + 16];
    acc[t] = __builtin_amdgcn_wmma_f32_16x16x32_bf16(
        false, aQ.v, false, bK.v, (short)0, acc[t], false, false);
  }

  // ---- add combined bias+mask (padding already holds -1e30) ----
  #pragma unroll
  for (int t = 0; t < 7; ++t)
    acc[t] += *(const v8f*)(cb + t * 256);

  // ---- f32 softmax over k (rows striped per C layout) ----
  float rmax[8], rsum[8];
  #pragma unroll
  for (int e = 0; e < 8; ++e) {
    float m = acc[0][e];
    #pragma unroll
    for (int t = 1; t < 7; ++t) m = fmaxf(m, acc[t][e]);
    rmax[e] = m;
  }
  #pragma unroll
  for (int off = 1; off < 16; off <<= 1) {
    #pragma unroll
    for (int e = 0; e < 8; ++e)
      rmax[e] = fmaxf(rmax[e], __shfl_xor(rmax[e], off, 32));
  }
  #pragma unroll
  for (int e = 0; e < 8; ++e) rsum[e] = 0.f;
  #pragma unroll
  for (int t = 0; t < 7; ++t) {
    #pragma unroll
    for (int e = 0; e < 8; ++e) {
      float p = __expf(acc[t][e] - rmax[e]);
      acc[t][e] = p;
      rsum[e] += p;
    }
  }
  #pragma unroll
  for (int off = 1; off < 16; off <<= 1) {
    #pragma unroll
    for (int e = 0; e < 8; ++e)
      rsum[e] += __shfl_xor(rsum[e], off, 32);
  }
  v8f rsv;
  #pragma unroll
  for (int e = 0; e < 8; ++e) rsv[e] = 1.0f / rsum[e];

  // ---- spill normalized P (packed cvt to bf16) to this wave's LDS slab ----
  #pragma unroll
  for (int t = 0; t < 7; ++t) {
    v8bf pbv = __builtin_convertvector(acc[t] * rsv, v8bf);
    #pragma unroll
    for (int e = 0; e < 8; ++e)
      Ps[(16 * mi + e + 8 * s) * KPAD + 16 * t + n16] = pbv[e];
  }
  const __bf16 bz = (__bf16)0.0f;
  #pragma unroll
  for (int e = 0; e < 8; ++e)            // zero padded cols 112..127
    Ps[(16 * mi + e + 8 * s) * KPAD + 112 + n16] = bz;

  // ---- O = P V : 4 K-chunks x 2 N-tiles = 8 WMMA (same-wave data) ----
  v8f o0 = zero8, o1 = zero8;
  #pragma unroll
  for (int kt = 0; kt < 4; ++kt) {
    Frag16 aP;
    int abase = (16 * mi + n16) * KPAD + 32 * kt;
    aP.u[0] = *(const uint4*)&Ps[abase + 8 * s];
    aP.u[1] = *(const uint4*)&Ps[abase + 16 + 8 * s];
    Frag16 bV0, bV1;
    int vb0 = n16 * KPAD + 32 * kt + 8 * s;
    int vb1 = (16 + n16) * KPAD + 32 * kt + 8 * s;
    bV0.u[0] = *(const uint4*)&Vts[vb0];
    bV0.u[1] = *(const uint4*)&Vts[vb0 + 16];
    bV1.u[0] = *(const uint4*)&Vts[vb1];
    bV1.u[1] = *(const uint4*)&Vts[vb1 + 16];
    o0 = __builtin_amdgcn_wmma_f32_16x16x32_bf16(
        false, aP.v, false, bV0.v, (short)0, o0, false, false);
    o1 = __builtin_amdgcn_wmma_f32_16x16x32_bf16(
        false, aP.v, false, bV1.v, (short)0, o1, false, false);
  }

  // ---- store O rows (q < 98) in head-concat (B_, 98, 128) layout ----
  float* Ob = Obuf + ((size_t)b * NTOK) * DIMC + h * HD;
  #pragma unroll
  for (int e = 0; e < 8; ++e) {
    int q = 16 * mi + e + 8 * s;
    if (q < NTOK) {
      Ob[(size_t)q * DIMC + n16]      = o0[e];
      Ob[(size_t)q * DIMC + 16 + n16] = o1[e];
    }
  }
}

// ---------------------------------------------------------------
// Kernel 2: out = O @ W^T + b   (200704 x 128 x 128)
// block = 256 threads = 8 waves; each wave -> 16x128 output slab
// ---------------------------------------------------------------
__global__ void __launch_bounds__(256) proj_kernel(
    const float* __restrict__ O,   // (200704, 128)
    const float* __restrict__ W,   // (128, 128) row-major
    const float* __restrict__ pb,  // (128,)
    float* __restrict__ out) {     // (200704, 128)
  __shared__ __align__(16) __bf16 Wbf[DIMC * DIMC];  // bf16 W[n][k]

  const int tid = threadIdx.x;
  for (int i = tid; i < DIMC * DIMC / 4; i += 256) {
    BF4 wb;
    wb.v = __builtin_convertvector(*(const v4f*)(W + i * 4), v4bf);
    *(uint2*)&Wbf[i * 4] = wb.u;
  }
  __syncthreads();

  const int wave = tid >> 5;
  const int lane = tid & 31;
  const int n16  = lane & 15;
  const int s    = lane >> 4;
  const size_t m0 = ((size_t)blockIdx.x * 8 + wave) * 16;

  v8f zero8 = {0.f, 0.f, 0.f, 0.f, 0.f, 0.f, 0.f, 0.f};
  v8f acc[8];
  #pragma unroll
  for (int nt = 0; nt < 8; ++nt) acc[nt] = zero8;

  const float* Arow = O + (m0 + n16) * DIMC;
  #pragma unroll
  for (int kt = 0; kt < 4; ++kt) {
    // A fragment: two v8f loads -> packed cvt -> bf16 layout halves
    Frag16 aO;
    const float* base = Arow + 32 * kt;
    aO.h[0] = __builtin_convertvector(*(const v8f*)(base + 8 * s), v8bf);
    aO.h[1] = __builtin_convertvector(*(const v8f*)(base + 16 + 8 * s), v8bf);

    #pragma unroll
    for (int nt = 0; nt < 8; ++nt) {
      Frag16 bW;
      int wb = (16 * nt + n16) * DIMC + 32 * kt + 8 * s;
      bW.u[0] = *(const uint4*)&Wbf[wb];
      bW.u[1] = *(const uint4*)&Wbf[wb + 16];
      acc[nt] = __builtin_amdgcn_wmma_f32_16x16x32_bf16(
          false, aO.v, false, bW.v, (short)0, acc[nt], false, false);
    }
  }

  #pragma unroll
  for (int nt = 0; nt < 8; ++nt) {
    int col = 16 * nt + n16;
    float pbv = pb[col];
    #pragma unroll
    for (int e = 0; e < 8; ++e)
      out[(m0 + e + 8 * s) * DIMC + col] = acc[nt][e] + pbv;
  }
}

// ---------------------------------------------------------------
extern "C" void kernel_launch(void* const* d_in, const int* in_sizes, int n_in,
                              void* d_out, int out_size, void* d_ws, size_t ws_size,
                              hipStream_t stream) {
  const float* x     = (const float*)d_in[0];
  const int*   rpi   = (const int*)d_in[1];
  const float* mask  = (const float*)d_in[2];
  const float* table = (const float*)d_in[3];
  const float* pw    = (const float*)d_in[4];
  const float* pbias = (const float*)d_in[5];
  float* out = (float*)d_out;

  // workspace: O intermediate (102.8 MB) then combined bias+mask (12.8 MB)
  float* Obuf = (float*)d_ws;
  float* comb = (float*)((char*)d_ws + (size_t)BWIN * NTOK * DIMC * sizeof(float));

  hipLaunchKernelGGL(comb_kernel, dim3((COMB_ELEMS + 255) / 256), dim3(256), 0,
                     stream, rpi, table, mask, comb);
  hipLaunchKernelGGL(attn_kernel, dim3(BWIN, NHEADS), dim3(256), 0, stream,
                     x, comb, Obuf);
  hipLaunchKernelGGL(proj_kernel, dim3((BWIN * NTOK / 16) / 8), dim3(256), 0,
                     stream, Obuf, pw, pbias, out);
}
